// AtomAttentionDecoder_91027536871995
// MI455X (gfx1250) — matrix operand order repaired and checked by reading, not compiled, not run on hardware
//
#include <hip/hip_runtime.h>
#include <hip/hip_bf16.h>
#include <cstdint>
#include <cstddef>

// ---------------------------------------------------------------------------
// AtomAttentionDecoder for MI455X (gfx1250, wave32, WMMA).
//  * Banded attention: block mask == each 32-query block attends keys in
//    [32c-48, 32c+80). plm is read only inside the band, once, for all
//    3 blocks (16.8 MB instead of 768 MB -> removes the HBM bottleneck).
//  * AdaLN gates/shifts + output gates depend only on s=cl_skip -> plain
//    GEMMs computed per block.
//  * All GEMMs + attention: v_wmma_f32_16x16x32_bf16 (f32 accumulate).
//  * Fragment builds use contiguous k order: A elements 0..15 are
//    K=16h..16h+15, as are B elements -> every fragment
//    is two aligned 16B loads (ds_load_b128 / global_load_b128), no
//    per-element gathers, no exec-mask branching.
// ---------------------------------------------------------------------------

typedef __attribute__((ext_vector_type(16))) __bf16 v16bf;
typedef __attribute__((ext_vector_type(8)))  __bf16 v8bf;
typedef __attribute__((ext_vector_type(8)))  float  v8f;

#define N_TOKEN_C 512
#define N_ATOM_C  2048
#define C_ATOM_C  128
#define C_TOKEN_C 768
#define NHEAD_C   4
#define NBLK_C    3

__device__ inline __bf16 f2bf(float f) {
    union { float f; unsigned u; } v; v.f = f;
    unsigned r = v.u + 0x7fffu + ((v.u >> 16) & 1u);   // round-to-nearest-even
    return __builtin_bit_cast(__bf16, (unsigned short)(r >> 16));
}
__device__ inline float sigmf(float v) { return 1.0f / (1.0f + __expf(-v)); }

// Two contiguous 8x-bf16 (16B) loads -> one 16-element fragment.
__device__ inline v16bf frag2(const __bf16* lo, const __bf16* hi) {
    v8bf a = *(const v8bf*)lo;
    v8bf b = *(const v8bf*)hi;
    return __builtin_shufflevector(a, b, 0, 1, 2, 3, 4, 5, 6, 7,
                                   8, 9, 10, 11, 12, 13, 14, 15);
}

// ---------------------------------------------------------------------------
// Weight packing: W[K][N] f32 -> P[nt][kc][col16][k32] bf16 so the WMMA
// B-fragment for (nt, kc, col=l16) is 32 contiguous bf16 (two b128 loads).
// ---------------------------------------------------------------------------
__global__ __launch_bounds__(256)
void pack_w(const float* __restrict__ W, __bf16* __restrict__ P, int K, int N)
{
    int idx = blockIdx.x * 256 + threadIdx.x;
    if (idx >= K * N) return;
    int k = idx / N, n = idx - k * N;
    int nt = n >> 4, col = n & 15, kc = k >> 5, kk = k & 31;
    P[((size_t)(nt * (K >> 5) + kc) * 16 + col) * 32 + kk] = f2bf(W[idx]);
}

// ---------------------------------------------------------------------------
// Generic WMMA GEMM: C[M,N] = epi(A[M,K] @ B[K,N] (+bias)), B pre-packed bf16.
// 256 threads = 8 waves; workgroup tile = 16 rows x N cols.
// ---------------------------------------------------------------------------
enum { EPI_NONE = 0, EPI_SIG = 1, EPI_SILU = 2, EPI_MUL = 3, EPI_MAD = 4 };

template <int EPI>
__global__ __launch_bounds__(256)
void gemm_bf16(const float* __restrict__ A, const __bf16* __restrict__ Bp,
               const float* __restrict__ bias, float* __restrict__ C,
               const float* __restrict__ aux0, const float* __restrict__ aux1,
               int M, int N, int K)
{
    __shared__ __bf16 lds_a[16 * 32];
    const int m0   = blockIdx.x * 16;
    const int tid  = threadIdx.x;
    const int wave = tid >> 5, lane = tid & 31;
    const int half = lane >> 4, l16 = lane & 15;
    const int ntiles  = N >> 4;
    const int kchunks = K >> 5;

    v8f acc[2] = {};

    for (int kc = 0; kc < kchunks; ++kc) {
        for (int i = tid; i < 512; i += 256) {
            int r = i >> 5, c = i & 31;
            lds_a[i] = f2bf(A[(size_t)(m0 + r) * K + kc * 32 + c]);
        }
        __syncthreads();

        const __bf16* ap = &lds_a[l16 * 32];
        v16bf af = frag2(ap + 16 * half, ap + 16 * half + 8);

#pragma unroll
        for (int t = 0; t < 2; ++t) {
            int nt = wave + 8 * t;
            if (nt < ntiles) {
                const __bf16* bp =
                    Bp + ((size_t)(nt * kchunks + kc) * 16 + l16) * 32 + 16 * half;
                v16bf bfr = frag2(bp, bp + 8);
                acc[t] = __builtin_amdgcn_wmma_f32_16x16x32_bf16(
                    false, af, false, bfr, (short)0, acc[t], false, false);
            }
        }
        __syncthreads();
    }

#pragma unroll
    for (int t = 0; t < 2; ++t) {
        int nt = wave + 8 * t;
        if (nt >= ntiles) continue;
#pragma unroll
        for (int r = 0; r < 8; ++r) {
            int m = m0 + r + 8 * half;
            int n = 16 * nt + l16;
            size_t o = (size_t)m * N + n;
            float v = acc[t][r];
            if (bias) v += bias[n];
            if (EPI == EPI_SIG)  v = sigmf(v);
            if (EPI == EPI_SILU) v = v * sigmf(v);
            if (EPI == EPI_MUL)  v = aux0[o] * v;
            if (EPI == EPI_MAD)  v = aux1[o] + aux0[o] * v;
            C[o] = v;
        }
    }
}

// ---------------------------------------------------------------------------
// Token->atom scan + broadcast residual: a = ql_skip + tok_emb[idx]
// ---------------------------------------------------------------------------
__global__ void scan_kernel(const int* __restrict__ apt, int* __restrict__ starts,
                            int n_token)
{
    if (blockIdx.x == 0 && threadIdx.x == 0) {
        int s = 0;
        for (int t = 0; t < n_token; ++t) { starts[t] = s; s += apt[t]; }
        starts[n_token] = s;
    }
}

__global__ void bcast_kernel(const float* __restrict__ ql, const float* __restrict__ tok,
                             const int* __restrict__ starts, float* __restrict__ a)
{
    int t = blockIdx.x, c = threadIdx.x;           // 128 threads/block
    int s0 = starts[t], s1 = starts[t + 1];
    float tv = tok[(size_t)t * C_ATOM_C + c];
    for (int atom = s0; atom < s1; ++atom)
        a[(size_t)atom * C_ATOM_C + c] = ql[(size_t)atom * C_ATOM_C + c] + tv;
}

// ---------------------------------------------------------------------------
// Banded pair bias: pb[b][h][q][j] for key m = 32*(q>>5) - 48 + j, j<128.
// ---------------------------------------------------------------------------
__global__ __launch_bounds__(256)
void pair_bias_kernel(const float* __restrict__ plm,
                      const float* __restrict__ pg, const float* __restrict__ pbias,
                      const float* __restrict__ wp, float* __restrict__ pb, int n_atom)
{
    int idx = blockIdx.x * blockDim.x + threadIdx.x;
    if (idx >= n_atom * 128) return;
    int q = idx >> 7, j = idx & 127;
    int m = 32 * (q >> 5) - 48 + j;
    size_t stride_bh = (size_t)n_atom * 128;

    if (m < 0 || m >= n_atom) {
        for (int bh = 0; bh < NBLK_C * NHEAD_C; ++bh)
            pb[bh * stride_bh + (size_t)q * 128 + j] = -1e10f;
        return;
    }
    const float* row = plm + ((size_t)q * n_atom + m) * 16;
    float z[16], mu = 0.f, s2 = 0.f;
#pragma unroll
    for (int c = 0; c < 16; ++c) { z[c] = row[c]; mu += z[c]; s2 += z[c] * z[c]; }
    mu *= (1.f / 16.f);
    float rin = rsqrtf(s2 * (1.f / 16.f) - mu * mu + 1e-5f);
#pragma unroll
    for (int b = 0; b < NBLK_C; ++b) {
        float acc[NHEAD_C] = {};
#pragma unroll
        for (int c = 0; c < 16; ++c) {
            float y = (z[c] - mu) * rin * pg[b * 16 + c] + pbias[b * 16 + c];
#pragma unroll
            for (int h = 0; h < NHEAD_C; ++h)
                acc[h] += y * wp[(b * 16 + c) * NHEAD_C + h];
        }
#pragma unroll
        for (int h = 0; h < NHEAD_C; ++h)
            pb[(size_t)(b * NHEAD_C + h) * stride_bh + (size_t)q * 128 + j] = acc[h];
    }
}

// ---------------------------------------------------------------------------
// Row LayerNorm helpers (wave per row, 128 ch, 4 per lane).
// ---------------------------------------------------------------------------
__device__ inline void row_stats(float4 v, float& mu, float& rin) {
    float s  = v.x + v.y + v.z + v.w;
    float s2 = v.x * v.x + v.y * v.y + v.z * v.z + v.w * v.w;
#pragma unroll
    for (int off = 1; off < 32; off <<= 1) {
        s  += __shfl_xor(s,  off, 32);
        s2 += __shfl_xor(s2, off, 32);
    }
    mu = s * (1.f / 128.f);
    rin = rsqrtf(s2 * (1.f / 128.f) - mu * mu + 1e-5f);
}

// x = gate * LN(a) + shift
__global__ __launch_bounds__(256)
void ln_mod_kernel(const float* __restrict__ a, const float* __restrict__ gate,
                   const float* __restrict__ shift, float* __restrict__ x, int M)
{
    int row = blockIdx.x * 8 + (threadIdx.x >> 5);
    int lane = threadIdx.x & 31;
    if (row >= M) return;
    size_t base = (size_t)row * 128;
    float4 v = ((const float4*)(a + base))[lane];
    float mu, rin; row_stats(v, mu, rin);
    float4 g = ((const float4*)(gate + base))[lane];
    float4 s = ((const float4*)(shift + base))[lane];
    float4 o;
    o.x = g.x * ((v.x - mu) * rin) + s.x;
    o.y = g.y * ((v.y - mu) * rin) + s.y;
    o.z = g.z * ((v.z - mu) * rin) + s.z;
    o.w = g.w * ((v.w - mu) * rin) + s.w;
    ((float4*)(x + base))[lane] = o;
}

// out_a = LN(s)*sg_a[c]; out_t = LN(s)*sg_t[c]
__global__ __launch_bounds__(256)
void sln_kernel(const float* __restrict__ sIn, const float* __restrict__ sg_a,
                const float* __restrict__ sg_t, float* __restrict__ out_a,
                float* __restrict__ out_t, int M)
{
    int row = blockIdx.x * 8 + (threadIdx.x >> 5);
    int lane = threadIdx.x & 31;
    if (row >= M) return;
    size_t base = (size_t)row * 128;
    float4 v = ((const float4*)(sIn + base))[lane];
    float mu, rin; row_stats(v, mu, rin);
    float4 ga = ((const float4*)sg_a)[lane];
    float4 gt = ((const float4*)sg_t)[lane];
    float4 y; y.x = (v.x - mu) * rin; y.y = (v.y - mu) * rin;
    y.z = (v.z - mu) * rin; y.w = (v.w - mu) * rin;
    float4 oa, ot;
    oa.x = y.x * ga.x; oa.y = y.y * ga.y; oa.z = y.z * ga.z; oa.w = y.w * ga.w;
    ot.x = y.x * gt.x; ot.y = y.y * gt.y; ot.z = y.z * gt.z; ot.w = y.w * gt.w;
    ((float4*)(out_a + base))[lane] = oa;
    ((float4*)(out_t + base))[lane] = ot;
}

// ---------------------------------------------------------------------------
// Banded attention. One workgroup (8 waves) per 32-query block.
// wave = 2*h + mt; wave owns 16 queries of head h.
// Q/K/V windows staged once in LDS as bf16 (branch-free, mask-multiplied);
// all fragments are paired ds_load_b128; S and O via WMMA.
// Dynamic LDS: (32*128 + 2*128*128 + 8*16*128) bf16 = 104 KB.
// ---------------------------------------------------------------------------
__global__ __launch_bounds__(256)
void attn_kernel(const float* __restrict__ q, const float* __restrict__ k,
                 const float* __restrict__ v, const float* __restrict__ g,
                 const float* __restrict__ pb, float* __restrict__ go, int n_atom)
{
    extern __shared__ __bf16 smem[];
    __bf16* lds_q = smem;                   // [32 q ][128 ch]
    __bf16* lds_k = lds_q + 32 * 128;       // [128 key][128 ch]
    __bf16* lds_v = lds_k + 128 * 128;      // [128 ch][128 key] (transposed)
    __bf16* lds_p = lds_v + 128 * 128;      // [8 wave][16 q][128 key]

    const int qb = blockIdx.x, q0 = qb * 32, w0 = q0 - 48;
    const int tid = threadIdx.x, wave = tid >> 5, lane = tid & 31;
    const int half = lane >> 4, l16 = lane & 15;
    const int h = wave >> 1, mt = wave & 1;
    const float inv_sqrt = 0.17677669529663687f;  // 1/sqrt(32)

    // ---- stage Q (always in range) ----
    for (int i = tid; i < 32 * 32; i += 256) {
        int r = i >> 5, c4 = i & 31;
        float4 x4 = ((const float4*)(q + (size_t)(q0 + r) * 128))[c4];
        __bf16* d = &lds_q[r * 128 + c4 * 4];
        d[0] = f2bf(x4.x); d[1] = f2bf(x4.y); d[2] = f2bf(x4.z); d[3] = f2bf(x4.w);
    }
    // ---- stage K/V window, branch-free zeroing outside [0, n_atom) ----
    for (int i = tid; i < 128 * 32; i += 256) {
        int r = i >> 5, c4 = i & 31;
        int key = w0 + r;
        int kcl = key < 0 ? 0 : (key >= n_atom ? n_atom - 1 : key);
        float msk = (key >= 0 && key < n_atom) ? 1.f : 0.f;
        float4 kx = ((const float4*)(k + (size_t)kcl * 128))[c4];
        float4 vx = ((const float4*)(v + (size_t)kcl * 128))[c4];
        int c0 = c4 * 4;
        __bf16* dk = &lds_k[r * 128 + c0];
        dk[0] = f2bf(kx.x * msk); dk[1] = f2bf(kx.y * msk);
        dk[2] = f2bf(kx.z * msk); dk[3] = f2bf(kx.w * msk);
        lds_v[(c0 + 0) * 128 + r] = f2bf(vx.x * msk);
        lds_v[(c0 + 1) * 128 + r] = f2bf(vx.y * msk);
        lds_v[(c0 + 2) * 128 + r] = f2bf(vx.z * msk);
        lds_v[(c0 + 3) * 128 + r] = f2bf(vx.w * msk);
    }
    __syncthreads();

    // ---- S = Q K^T ----
    const __bf16* qrow = &lds_q[(16 * mt + l16) * 128 + 32 * h];
    v16bf qf = frag2(qrow + 16 * half, qrow + 16 * half + 8);

    v8f s[8];
#pragma unroll
    for (int nt = 0; nt < 8; ++nt) {
        const __bf16* kp = &lds_k[(16 * nt + l16) * 128 + 32 * h + 16 * half];
        v16bf kf = frag2(kp, kp + 8);
        v8f z = {};
        s[nt] = __builtin_amdgcn_wmma_f32_16x16x32_bf16(false, qf, false, kf,
                                                        (short)0, z, false, false);
    }

    // ---- softmax over the 128-key band (pb carries the -1e10 mask) ----
    const int rowbase = q0 + 16 * mt;
    float rowsum[8];
#pragma unroll
    for (int r = 0; r < 8; ++r) {
        int qrw = rowbase + r + 8 * half;
        float mx = -3.0e38f;
#pragma unroll
        for (int nt = 0; nt < 8; ++nt) {
            float lg = s[nt][r] * inv_sqrt +
                       pb[((size_t)h * n_atom + qrw) * 128 + 16 * nt + l16];
            s[nt][r] = lg;
            mx = fmaxf(mx, lg);
        }
#pragma unroll
        for (int off = 1; off < 16; off <<= 1) mx = fmaxf(mx, __shfl_xor(mx, off, 32));
        float sm = 0.f;
#pragma unroll
        for (int nt = 0; nt < 8; ++nt) {
            float p = __expf(s[nt][r] - mx);
            s[nt][r] = p;
            sm += p;
        }
#pragma unroll
        for (int off = 1; off < 16; off <<= 1) sm += __shfl_xor(sm, off, 32);
        rowsum[r] = sm;
    }

    // ---- P -> LDS (bf16, row-major 16x128 per wave) ----
    __bf16* pw = lds_p + wave * 16 * 128;
#pragma unroll
    for (int nt = 0; nt < 8; ++nt)
#pragma unroll
        for (int r = 0; r < 8; ++r)
            pw[(r + 8 * half) * 128 + 16 * nt + l16] = f2bf(s[nt][r]);
    __syncthreads();

    // ---- O = P V ----
#pragma unroll
    for (int ot = 0; ot < 2; ++ot) {
        v8f o = {};
#pragma unroll
        for (int kc = 0; kc < 4; ++kc) {
            const __bf16* pp = &pw[l16 * 128 + 32 * kc];
            v16bf pf = frag2(pp + 16 * half, pp + 16 * half + 8);
            const __bf16* vp = &lds_v[(32 * h + 16 * ot + l16) * 128 + 32 * kc + 16 * half];
            v16bf vf = frag2(vp, vp + 8);
            o = __builtin_amdgcn_wmma_f32_16x16x32_bf16(false, pf, false, vf,
                                                        (short)0, o, false, false);
        }
#pragma unroll
        for (int r = 0; r < 8; ++r) {
            int qrw = rowbase + r + 8 * half;
            int n = 32 * h + 16 * ot + l16;
            size_t off = (size_t)qrw * 128 + n;
            go[off] = (o[r] / rowsum[r]) * g[off];
        }
    }
}

// ---------------------------------------------------------------------------
// Final: out[M,3] = LN(a; g, b) @ w_out   (N=3: wave per row)
// ---------------------------------------------------------------------------
__global__ __launch_bounds__(256)
void final_kernel(const float* __restrict__ a, const float* __restrict__ g,
                  const float* __restrict__ b, const float* __restrict__ w_out,
                  float* __restrict__ out, int M)
{
    int row = blockIdx.x * 8 + (threadIdx.x >> 5);
    int lane = threadIdx.x & 31;
    if (row >= M) return;
    float4 v = ((const float4*)(a + (size_t)row * 128))[lane];
    float mu, rin; row_stats(v, mu, rin);
    float vv[4] = { v.x, v.y, v.z, v.w };
    float acc0 = 0.f, acc1 = 0.f, acc2 = 0.f;
#pragma unroll
    for (int i = 0; i < 4; ++i) {
        int c = lane * 4 + i;
        float y = (vv[i] - mu) * rin * g[c] + b[c];
        acc0 += y * w_out[c * 3 + 0];
        acc1 += y * w_out[c * 3 + 1];
        acc2 += y * w_out[c * 3 + 2];
    }
#pragma unroll
    for (int off = 1; off < 32; off <<= 1) {
        acc0 += __shfl_xor(acc0, off, 32);
        acc1 += __shfl_xor(acc1, off, 32);
        acc2 += __shfl_xor(acc2, off, 32);
    }
    if (lane == 0) {
        out[row * 3 + 0] = acc0;
        out[row * 3 + 1] = acc1;
        out[row * 3 + 2] = acc2;
    }
}

// ---------------------------------------------------------------------------
// Host orchestration. Input order: setup_inputs() insertion order, then the
// params pytree flattened in jax sorted-key order:
//   0 ai  1 ql_skip  2 cl_skip  3 plm  4 atom_per_token
//   5..28 blocks.{a_gb,a_gw,a_lw,a_ogb,a_ogw,a_sg,bq,pair_ln_b,pair_ln_g,
//                 t_gb,t_gw,t_lw,t_ogb,t_ogw,t_sg,w1,w2,w3,w_pair,wg,wk,wo,wq,wv}
//   29 ln_b  30 ln_g  31 w_in  32 w_out
// ---------------------------------------------------------------------------
extern "C" void kernel_launch(void* const* d_in, const int* in_sizes, int n_in,
                              void* d_out, int out_size, void* d_ws, size_t ws_size,
                              hipStream_t stream)
{
    (void)in_sizes; (void)n_in; (void)out_size; (void)ws_size;
    const float* ai      = (const float*)d_in[0];
    const float* ql_skip = (const float*)d_in[1];
    const float* cl_skip = (const float*)d_in[2];
    const float* plm     = (const float*)d_in[3];
    const int*   apt     = (const int*)d_in[4];
    const float* a_gb    = (const float*)d_in[5];
    const float* a_gw    = (const float*)d_in[6];
    const float* a_lw    = (const float*)d_in[7];
    const float* a_ogb   = (const float*)d_in[8];
    const float* a_ogw   = (const float*)d_in[9];
    const float* a_sg    = (const float*)d_in[10];
    const float* bq      = (const float*)d_in[11];
    const float* pln_b   = (const float*)d_in[12];
    const float* pln_g   = (const float*)d_in[13];
    const float* t_gb    = (const float*)d_in[14];
    const float* t_gw    = (const float*)d_in[15];
    const float* t_lw    = (const float*)d_in[16];
    const float* t_ogb   = (const float*)d_in[17];
    const float* t_ogw   = (const float*)d_in[18];
    const float* t_sg    = (const float*)d_in[19];
    const float* w1      = (const float*)d_in[20];
    const float* w2      = (const float*)d_in[21];
    const float* w3      = (const float*)d_in[22];
    const float* w_pair  = (const float*)d_in[23];
    const float* wg      = (const float*)d_in[24];
    const float* wk      = (const float*)d_in[25];
    const float* wo      = (const float*)d_in[26];
    const float* wq      = (const float*)d_in[27];
    const float* wv      = (const float*)d_in[28];
    const float* ln_b    = (const float*)d_in[29];
    const float* ln_g    = (const float*)d_in[30];
    const float* w_in    = (const float*)d_in[31];
    const float* w_out   = (const float*)d_in[32];

    const int M = N_ATOM_C, C = C_ATOM_C;
    const size_t MC  = (size_t)M * C;
    const size_t MC2 = (size_t)M * 2 * C;
    const size_t CC  = (size_t)C * C;       // 16384
    const size_t C2C = (size_t)C * 2 * C;   // 32768

    char* Wb = (char*)d_ws;
    size_t off = 0;
    auto allocf = [&](size_t n) { float* p = (float*)(Wb + off); off += n * 4; return p; };
    auto allocb = [&](size_t n) { __bf16* p = (__bf16*)(Wb + off); off += ((n * 2 + 15) & ~(size_t)15); return p; };

    float* tok   = allocf((size_t)N_TOKEN_C * C);
    float* a     = allocf(MC);
    float* x     = allocf(MC);
    float* y     = allocf(MC);
    float* qb_   = allocf(MC);
    float* kb_   = allocf(MC);
    float* vb_   = allocf(MC);
    float* gb_   = allocf(MC);
    float* go    = allocf(MC);
    float* anew  = allocf(MC);
    float* sln_a = allocf(MC);
    float* sln_t = allocf(MC);
    float* ga    = allocf(MC);
    float* sa    = allocf(MC);
    float* gt    = allocf(MC);
    float* st    = allocf(MC);
    float* oga   = allocf(MC);
    float* ogt   = allocf(MC);
    float* t1    = allocf(MC2);
    float* h1    = allocf(MC2);
    float* pb    = allocf((size_t)NBLK_C * NHEAD_C * M * 128);
    int*   starts = (int*)allocf(N_TOKEN_C + 4);

    // packed bf16 weights
    __bf16* P_win = allocb((size_t)C_TOKEN_C * C);
    __bf16* P_agw = allocb(NBLK_C * CC);
    __bf16* P_alw = allocb(NBLK_C * CC);
    __bf16* P_tgw = allocb(NBLK_C * CC);
    __bf16* P_tlw = allocb(NBLK_C * CC);
    __bf16* P_aog = allocb(NBLK_C * CC);
    __bf16* P_tog = allocb(NBLK_C * CC);
    __bf16* P_wq  = allocb(NBLK_C * CC);
    __bf16* P_wk  = allocb(NBLK_C * CC);
    __bf16* P_wv  = allocb(NBLK_C * CC);
    __bf16* P_wg  = allocb(NBLK_C * CC);
    __bf16* P_wo  = allocb(NBLK_C * CC);
    __bf16* P_w1  = allocb(NBLK_C * C2C);
    __bf16* P_w2  = allocb(NBLK_C * C2C);
    __bf16* P_w3  = allocb(NBLK_C * C2C);

    dim3 B256(256);
    dim3 gM(M / 16), gTok(N_TOKEN_C / 16), gRow(M / 8);

    auto pack = [&](const float* src, __bf16* dst, int K, int N) {
        int total = K * N;
        pack_w<<<dim3((total + 255) / 256), B256, 0, stream>>>(src, dst, K, N);
    };

    pack(w_in, P_win, C_TOKEN_C, C);
    for (int b = 0; b < NBLK_C; ++b) {
        pack(a_gw  + b * CC,  P_agw + b * CC,  C, C);
        pack(a_lw  + b * CC,  P_alw + b * CC,  C, C);
        pack(t_gw  + b * CC,  P_tgw + b * CC,  C, C);
        pack(t_lw  + b * CC,  P_tlw + b * CC,  C, C);
        pack(a_ogw + b * CC,  P_aog + b * CC,  C, C);
        pack(t_ogw + b * CC,  P_tog + b * CC,  C, C);
        pack(wq    + b * CC,  P_wq  + b * CC,  C, C);
        pack(wk    + b * CC,  P_wk  + b * CC,  C, C);
        pack(wv    + b * CC,  P_wv  + b * CC,  C, C);
        pack(wg    + b * CC,  P_wg  + b * CC,  C, C);
        pack(wo    + b * CC,  P_wo  + b * CC,  C, C);
        pack(w1    + b * C2C, P_w1  + b * C2C, C, 2 * C);
        pack(w2    + b * C2C, P_w2  + b * C2C, C, 2 * C);
        pack(w3    + b * C2C, P_w3  + b * C2C, 2 * C, C);
    }

    // a = ql_skip + (ai @ w_in)[idx]
    gemm_bf16<EPI_NONE><<<gTok, B256, 0, stream>>>(ai, P_win, nullptr, tok,
                                                   nullptr, nullptr,
                                                   N_TOKEN_C, C, C_TOKEN_C);
    scan_kernel<<<1, 32, 0, stream>>>(apt, starts, N_TOKEN_C);
    bcast_kernel<<<dim3(N_TOKEN_C), dim3(C), 0, stream>>>(ql_skip, tok, starts, a);

    // banded pair bias for all 3 blocks / 4 heads, single plm pass
    pair_bias_kernel<<<dim3((M * 128) / 256), B256, 0, stream>>>(
        plm, pln_g, pln_b, w_pair, pb, M);

    const size_t attn_lds = (size_t)(32 * 128 + 2 * 128 * 128 + 8 * 16 * 128) * 2;

    for (int b = 0; b < NBLK_C; ++b) {
        const size_t cc = b * CC, cv = (size_t)b * C, c2 = b * C2C;
        // conditioning (depends only on s = cl_skip)
        sln_kernel<<<gRow, B256, 0, stream>>>(cl_skip, a_sg + cv, t_sg + cv,
                                              sln_a, sln_t, M);
        gemm_bf16<EPI_SIG ><<<gM, B256, 0, stream>>>(sln_a, P_agw + cc, a_gb + cv, ga, nullptr, nullptr, M, C, C);
        gemm_bf16<EPI_NONE><<<gM, B256, 0, stream>>>(sln_a, P_alw + cc, nullptr,   sa, nullptr, nullptr, M, C, C);
        gemm_bf16<EPI_SIG ><<<gM, B256, 0, stream>>>(sln_t, P_tgw + cc, t_gb + cv, gt, nullptr, nullptr, M, C, C);
        gemm_bf16<EPI_NONE><<<gM, B256, 0, stream>>>(sln_t, P_tlw + cc, nullptr,   st, nullptr, nullptr, M, C, C);
        gemm_bf16<EPI_SIG ><<<gM, B256, 0, stream>>>(cl_skip, P_aog + cc, a_ogb + cv, oga, nullptr, nullptr, M, C, C);
        gemm_bf16<EPI_SIG ><<<gM, B256, 0, stream>>>(cl_skip, P_tog + cc, t_ogb + cv, ogt, nullptr, nullptr, M, C, C);

        // x = AdaLN(a, s); q/k/v/g projections
        ln_mod_kernel<<<gRow, B256, 0, stream>>>(a, ga, sa, x, M);
        gemm_bf16<EPI_NONE><<<gM, B256, 0, stream>>>(x, P_wq + cc, bq + cv, qb_, nullptr, nullptr, M, C, C);
        gemm_bf16<EPI_NONE><<<gM, B256, 0, stream>>>(x, P_wk + cc, nullptr, kb_, nullptr, nullptr, M, C, C);
        gemm_bf16<EPI_NONE><<<gM, B256, 0, stream>>>(x, P_wv + cc, nullptr, vb_, nullptr, nullptr, M, C, C);
        gemm_bf16<EPI_SIG ><<<gM, B256, 0, stream>>>(x, P_wg + cc, nullptr, gb_, nullptr, nullptr, M, C, C);

        // banded attention, gated output
        attn_kernel<<<dim3(M / 32), B256, attn_lds, stream>>>(
            qb_, kb_, vb_, gb_, pb + (size_t)b * NHEAD_C * M * 128, go, M);
        // attn_out = oga * (go @ wo)
        gemm_bf16<EPI_MUL><<<gM, B256, 0, stream>>>(go, P_wo + cc, nullptr, anew, oga, nullptr, M, C, C);

        // ConditionedTransitionBlock on pre-attention a
        ln_mod_kernel<<<gRow, B256, 0, stream>>>(a, gt, st, y, M);
        gemm_bf16<EPI_SILU><<<gM, B256, 0, stream>>>(y, P_w1 + c2, nullptr, t1, nullptr, nullptr, M, 2 * C, C);
        gemm_bf16<EPI_MUL ><<<gM, B256, 0, stream>>>(y, P_w2 + c2, nullptr, h1, t1, nullptr, M, 2 * C, C);
        // a = attn_out + ogt * (h1 @ w3)
        gemm_bf16<EPI_MAD ><<<gM, B256, 0, stream>>>(h1, P_w3 + c2, nullptr, a, ogt, anew, M, C, 2 * C);
    }

    final_kernel<<<gRow, B256, 0, stream>>>(a, ln_g, ln_b, w_out, (float*)d_out, M);
}
